// TransformerLM_5308579577949
// MI455X (gfx1250) — matrix operand order repaired and checked by reading, not compiled, not run on hardware
//
#include <hip/hip_runtime.h>
#include <hip/hip_bf16.h>

// ---------------------------------------------------------------------------
// MoE transformer forward for MI455X (gfx1250), f16 WMMA + fp32 accumulate.
// All matmuls use v_wmma_f32_16x16x32_f16 with [N][K]-major B operands so
// every fragment load is a contiguous b128. Each wave computes a 32x32 tile
// (4 WMMAs per 8 b128 loads); a 4-wave block covers 64x64.
// ---------------------------------------------------------------------------

typedef _Float16 half_t;
typedef __attribute__((ext_vector_type(16))) _Float16 v16h;
typedef __attribute__((ext_vector_type(8)))  _Float16 v8h;
typedef __attribute__((ext_vector_type(8)))  float    v8f;

constexpr int Dc = 512, Hc = 8, Ec = 4, Lc = 6, FFc = 2048, Sc = 1024, Bc = 2;
constexpr int HDc = Dc / Hc;          // 64
constexpr int Tc  = Sc * Bc;          // 2048 tokens
constexpr int D3c = 3 * Dc;           // 1536
constexpr int BHc = Bc * Hc;          // 16

// ------------------------------ generic WMMA GEMM ---------------------------
// C[M,N] = epilogue( alpha * A[M,K](f16) * B[N][K](f16)^T )
// batch z split as z = z1*Z0 + z0 with independent strides per operand.
struct GemmP {
  int M, N, K;
  const half_t* A; long lda, a0, a1;
  const half_t* B; long ldb, b0, b1;
  int Z0;
  const float* bias; long bias0, bias1;            // bias[n], nullable
  const float* rowScale; long rs0, rs1; int rsRow; // per-row scale, nullable
  float alpha; int causal; int relu;
  half_t* out16; float* out32; int accum;          // either, nullable
  long ldc, c0, c1;
};

__global__ __launch_bounds__(128) void wmma_gemm(GemmP p) {
  const int z  = blockIdx.z;
  const int z0 = (p.Z0 > 1) ? (z % p.Z0) : 0;
  const int z1 = (p.Z0 > 1) ? (z / p.Z0) : z;
  const int wid = threadIdx.y;                       // 4 waves in 2x2
  const int n0 = (blockIdx.x * 2 + (wid & 1)) * 32;
  const int m0 = (blockIdx.y * 2 + (wid >> 1)) * 32;
  const int lane = threadIdx.x & 31;

  const half_t* Ab = p.A + (long)z1 * p.a1 + (long)z0 * p.a0 + (long)m0 * p.lda;
  const half_t* Bb = p.B + (long)z1 * p.b1 + (long)z0 * p.b0;

  // Fragment striping used here, same k order for A and B (wave32): A rows across lanes with 16 contiguous K per lane;
  // B columns across lanes with 16 contiguous K per lane.
  const int am = lane & 15, ak = (lane >> 4) * 16;
  const int bn = lane & 15, bk = (lane >> 4) * 16;
  const half_t* arow0 = Ab + (long)am * p.lda + ak;
  const half_t* arow1 = arow0 + 16 * p.lda;
  const half_t* brow0 = Bb + (long)(n0 + bn) * p.ldb + bk;
  const half_t* brow1 = brow0 + 16 * p.ldb;

  v8f acc00 = {}, acc01 = {}, acc10 = {}, acc11 = {};
  for (int k0 = 0; k0 < p.K; k0 += 32) {
    const v8h al0 = *(const v8h*)(arow0 + k0);
    const v8h ah0 = *(const v8h*)(arow0 + k0 + 8);
    const v8h al1 = *(const v8h*)(arow1 + k0);
    const v8h ah1 = *(const v8h*)(arow1 + k0 + 8);
    const v16h b0 = *(const v16h*)(brow0 + k0);          // 32B contiguous
    const v16h b1 = *(const v16h*)(brow1 + k0);
    __builtin_prefetch(brow0 + k0 + 256, 0, 3);          // global_prefetch_b8
    __builtin_prefetch(arow0 + k0 + 256, 0, 3);
    const v16h a0 = __builtin_shufflevector(al0, ah0, 0, 1, 2, 3, 4, 5, 6, 7,
                                            8, 9, 10, 11, 12, 13, 14, 15);
    const v16h a1 = __builtin_shufflevector(al1, ah1, 0, 1, 2, 3, 4, 5, 6, 7,
                                            8, 9, 10, 11, 12, 13, 14, 15);
    acc00 = __builtin_amdgcn_wmma_f32_16x16x32_f16(false, a0, false, b0,
                                                   (short)0, acc00, false, false);
    acc01 = __builtin_amdgcn_wmma_f32_16x16x32_f16(false, a0, false, b1,
                                                   (short)0, acc01, false, false);
    acc10 = __builtin_amdgcn_wmma_f32_16x16x32_f16(false, a1, false, b0,
                                                   (short)0, acc10, false, false);
    acc11 = __builtin_amdgcn_wmma_f32_16x16x32_f16(false, a1, false, b1,
                                                   (short)0, acc11, false, false);
  }

  // Epilogue: C/D layout -> VGPR r: lanes0-15 M=r, lanes16-31 M=r+8, N=lane&15
  const int half_id = lane >> 4;
  const long cOff = (long)z1 * p.c1 + (long)z0 * p.c0;
  const float* biasP = p.bias ? p.bias + (long)z1 * p.bias1 + (long)z0 * p.bias0 : nullptr;
  const float* rsP = p.rowScale ? p.rowScale + (long)z1 * p.rs1 + (long)z0 * p.rs0 : nullptr;
  const float bval[2] = { biasP ? biasP[n0 + (lane & 15)] : 0.0f,
                          biasP ? biasP[n0 + 16 + (lane & 15)] : 0.0f };
#pragma unroll
  for (int t = 0; t < 2; ++t) {
#pragma unroll
    for (int u = 0; u < 2; ++u) {
      const v8f& acc = t ? (u ? acc11 : acc10) : (u ? acc01 : acc00);
      const int gn = n0 + u * 16 + (lane & 15);
#pragma unroll
      for (int r = 0; r < 8; ++r) {
        const int gm = m0 + t * 16 + r + half_id * 8;
        float v = acc[r] * p.alpha + bval[u];
        if (p.causal && gn > gm) v += -1e9f;
        if (p.relu) v = fmaxf(v, 0.0f);
        if (rsP) v *= rsP[(long)gm * p.rsRow];
        const long idx = cOff + (long)gm * p.ldc + gn;
        if (p.out16) p.out16[idx] = (half_t)v;
        if (p.out32) { if (p.accum) p.out32[idx] += v; else p.out32[idx] = v; }
      }
    }
  }
}

// ------------------------------ elementwise/reduction kernels ---------------

__global__ void cvt_kernel(const float* __restrict__ in, half_t* o16, float* o32, long n) {
  long i = (long)blockIdx.x * blockDim.x + threadIdx.x;
  const long stride = (long)gridDim.x * blockDim.x;
  for (; i < n; i += stride) {
    const float v = in[i];
    if (o16) o16[i] = (half_t)v;
    if (o32) o32[i] = v;
  }
}

// vt[e][b][h][hd][s] = qkv[e][(s*B+b)*3D + 2D + h*HD + hd]  (V pre-transpose
// so the attention AV GEMM gets contiguous [N][K] fragments).
__global__ void transpose_v(const half_t* __restrict__ qkv, half_t* __restrict__ vt) {
  const long total = (long)Ec * BHc * HDc * Sc;
  long idx = (long)blockIdx.x * blockDim.x + threadIdx.x;
  if (idx >= total) return;
  const int s  = (int)(idx & (Sc - 1));
  long r = idx >> 10;                 // / Sc
  const int hd = (int)(r & (HDc - 1));
  r >>= 6;                            // / HDc
  const int h  = (int)(r & (Hc - 1));
  r >>= 3;                            // / Hc
  const int b  = (int)(r & (Bc - 1));
  const int e  = (int)(r >> 1);
  vt[idx] = qkv[((long)e * Tc + (long)s * Bc + b) * D3c + 2 * Dc + h * HDc + hd];
}

// Per-token expert gate: softmax(x @ Wg) + aux-loss stats (count of argmax, sum g).
__global__ void gate_kernel(const float* __restrict__ x, const float* __restrict__ Wg,
                            float* __restrict__ g, float* __restrict__ auxbuf) {
  const int t = blockIdx.x * blockDim.x + threadIdx.x;
  if (t >= Tc) return;
  const float* xr = x + (long)t * Dc;
  float a0 = 0.f, a1 = 0.f, a2 = 0.f, a3 = 0.f;
  for (int d = 0; d < Dc; ++d) {
    const float xv = xr[d];
    const float4 wv = *(const float4*)(Wg + (long)d * Ec);
    a0 += xv * wv.x; a1 += xv * wv.y; a2 += xv * wv.z; a3 += xv * wv.w;
  }
  float lg[Ec] = {a0, a1, a2, a3};
  int amax = 0; float best = lg[0];
  for (int e = 1; e < Ec; ++e) if (lg[e] > best) { best = lg[e]; amax = e; }
  float s = 0.f, pr[Ec];
  for (int e = 0; e < Ec; ++e) { pr[e] = expf(lg[e] - best); s += pr[e]; }
  const float inv = 1.f / s;
  for (int e = 0; e < Ec; ++e) {
    pr[e] *= inv;
    g[(long)t * Ec + e] = pr[e];
    atomicAdd(&auxbuf[Ec + e], pr[e]);
  }
  atomicAdd(&auxbuf[amax], 1.0f);
}

__global__ void aux_finalize(const float* __restrict__ auxbuf, float* __restrict__ outAux,
                             float invT) {
  if (threadIdx.x == 0 && blockIdx.x == 0) {
    float s = 0.f;
    for (int e = 0; e < Ec; ++e) s += (auxbuf[e] * invT) * (auxbuf[Ec + e] * invT);
    *outAux += (float)Ec * s;
  }
}

// Row softmax over `cols` fp32 scores -> f16 probabilities. One block per row.
__global__ void softmax_rows(const float* __restrict__ in, half_t* __restrict__ out, int cols) {
  __shared__ float red[256];
  const long base = (long)blockIdx.x * cols;
  const int tid = threadIdx.x;
  float vmax = -3.4e38f;
  for (int c = tid; c < cols; c += 256) vmax = fmaxf(vmax, in[base + c]);
  red[tid] = vmax; __syncthreads();
  for (int s = 128; s > 0; s >>= 1) { if (tid < s) red[tid] = fmaxf(red[tid], red[tid + s]); __syncthreads(); }
  const float m = red[0]; __syncthreads();
  float sum = 0.f;
  for (int c = tid; c < cols; c += 256) sum += expf(in[base + c] - m);
  red[tid] = sum; __syncthreads();
  for (int s = 128; s > 0; s >>= 1) { if (tid < s) red[tid] += red[tid + s]; __syncthreads(); }
  const float inv = 1.f / red[0];
  for (int c = tid; c < cols; c += 256) out[base + c] = (half_t)(expf(in[base + c] - m) * inv);
}

// y = LN(a (+ b)) * g + beta, over 512 cols; 256 threads = 2 elems/thread.
__global__ void ln_kernel(const float* __restrict__ a, const float* __restrict__ b,
                          const float* __restrict__ g, const float* __restrict__ be,
                          float* __restrict__ o32, half_t* o16) {
  __shared__ float red[256];
  const long base = (long)blockIdx.x * Dc;
  const int tid = threadIdx.x;
  float v0 = a[base + tid];
  float v1 = a[base + tid + 256];
  if (b) { v0 += b[base + tid]; v1 += b[base + tid + 256]; }
  red[tid] = v0 + v1; __syncthreads();
  for (int s = 128; s > 0; s >>= 1) { if (tid < s) red[tid] += red[tid + s]; __syncthreads(); }
  const float mean = red[0] * (1.0f / Dc); __syncthreads();
  const float d0 = v0 - mean, d1 = v1 - mean;
  red[tid] = d0 * d0 + d1 * d1; __syncthreads();
  for (int s = 128; s > 0; s >>= 1) { if (tid < s) red[tid] += red[tid + s]; __syncthreads(); }
  const float rstd = rsqrtf(red[0] * (1.0f / Dc) + 1e-5f);
  const float y0 = d0 * rstd * g[tid] + be[tid];
  const float y1 = d1 * rstd * g[tid + 256] + be[tid + 256];
  o32[base + tid] = y0; o32[base + tid + 256] = y1;
  if (o16) { o16[base + tid] = (half_t)y0; o16[base + tid + 256] = (half_t)y1; }
}

// ------------------------------ host orchestration --------------------------

static inline void launch_gemm(const GemmP& p, int batch, hipStream_t s) {
  dim3 grid(p.N / 64, p.M / 64, batch);
  hipLaunchKernelGGL(wmma_gemm, grid, dim3(32, 4), 0, s, p);
}

extern "C" void kernel_launch(void* const* d_in, const int* in_sizes, int n_in,
                              void* d_out, int out_size, void* d_ws, size_t ws_size,
                              hipStream_t stream) {
  (void)in_sizes; (void)n_in; (void)out_size; (void)ws_size;
  const float* src  = (const float*)d_in[0];
  const float* Wg   = (const float*)d_in[1];
  const float* Wqkv = (const float*)d_in[2];
  const float* bqkv = (const float*)d_in[3];
  const float* Wo   = (const float*)d_in[4];
  const float* bo   = (const float*)d_in[5];
  const float* W1   = (const float*)d_in[6];
  const float* b1   = (const float*)d_in[7];
  const float* W2   = (const float*)d_in[8];
  const float* b2   = (const float*)d_in[9];
  const float* ln1g = (const float*)d_in[10];
  const float* ln1b = (const float*)d_in[11];
  const float* ln2g = (const float*)d_in[12];
  const float* ln2b = (const float*)d_in[13];
  const float* ln3g = (const float*)d_in[14];
  const float* ln3b = (const float*)d_in[15];
  const float* lnfg = (const float*)d_in[16];
  const float* lnfb = (const float*)d_in[17];
  float* out = (float*)d_out;

  // workspace carve-out (256B aligned)
  char* w = (char*)d_ws;
  auto alloc = [&](size_t bytes) { char* p = w; w += (bytes + 255) & ~(size_t)255; return p; };
  half_t* Wqkvh = (half_t*)alloc((size_t)Lc * Ec * D3c * Dc * 2);
  half_t* Woh   = (half_t*)alloc((size_t)Lc * Ec * Dc * Dc * 2);
  half_t* W1h   = (half_t*)alloc((size_t)Lc * FFc * Dc * 2);
  half_t* W2h   = (half_t*)alloc((size_t)Lc * Dc * FFc * 2);
  half_t* xh    = (half_t*)alloc((size_t)Tc * Dc * 2);
  half_t* qkvh  = (half_t*)alloc((size_t)Ec * Tc * D3c * 2);
  half_t* vtAll = (half_t*)alloc((size_t)Ec * BHc * HDc * Sc * 2);
  half_t* oh    = (half_t*)alloc((size_t)Tc * Dc * 2);
  half_t* hh    = (half_t*)alloc((size_t)Tc * FFc * 2);
  half_t* probs = (half_t*)alloc((size_t)BHc * Sc * Sc * 2);
  float*  x32   = (float*)alloc((size_t)Tc * Dc * 4);
  float*  attn32= (float*)alloc((size_t)Tc * Dc * 4);
  float*  ff32  = (float*)alloc((size_t)Tc * Dc * 4);
  float*  g32   = (float*)alloc((size_t)Tc * Ec * 4);
  float*  sc32  = (float*)alloc((size_t)BHc * Sc * Sc * 4);
  float*  auxbuf= (float*)alloc(2 * Ec * 4);

  // fp32 -> f16 weight conversion (f16 weights ~75MB: resident in 192MB L2)
  cvt_kernel<<<4096, 256, 0, stream>>>(Wqkv, Wqkvh, nullptr, (long)Lc * Ec * D3c * Dc);
  cvt_kernel<<<2048, 256, 0, stream>>>(Wo,   Woh,   nullptr, (long)Lc * Ec * Dc * Dc);
  cvt_kernel<<<2048, 256, 0, stream>>>(W1,   W1h,   nullptr, (long)Lc * FFc * Dc);
  cvt_kernel<<<2048, 256, 0, stream>>>(W2,   W2h,   nullptr, (long)Lc * Dc * FFc);
  cvt_kernel<<<2048, 256, 0, stream>>>(src,  xh,    x32,     (long)Tc * Dc);
  hipMemsetAsync(out + (size_t)Tc * Dc, 0, sizeof(float), stream);  // aux accumulator

  for (int l = 0; l < Lc; ++l) {
    // ---- expert gate + aux loss
    hipMemsetAsync(auxbuf, 0, 2 * Ec * sizeof(float), stream);
    gate_kernel<<<(Tc + 255) / 256, 256, 0, stream>>>(x32, Wg + (size_t)l * Dc * Ec, g32, auxbuf);
    aux_finalize<<<1, 1, 0, stream>>>(auxbuf, out + (size_t)Tc * Dc, 1.0f / Tc);

    // ---- qkv = x @ Wqkv[e]^T + bqkv[e]   (batched over experts)
    { GemmP p{};
      p.M = Tc; p.N = D3c; p.K = Dc;
      p.A = xh; p.lda = Dc; p.a0 = 0; p.a1 = 0;
      p.B = Wqkvh + (size_t)l * Ec * D3c * Dc; p.ldb = Dc; p.b0 = 0; p.b1 = (long)D3c * Dc;
      p.Z0 = 1;
      p.bias = bqkv + (size_t)l * Ec * D3c; p.bias0 = 0; p.bias1 = D3c;
      p.alpha = 1.f;
      p.out16 = qkvh; p.ldc = D3c; p.c0 = 0; p.c1 = (long)Tc * D3c;
      launch_gemm(p, Ec, stream);
    }
    // ---- pre-transpose V for the AV GEMM
    transpose_v<<<(int)(((long)Ec * BHc * HDc * Sc + 255) / 256), 256, 0, stream>>>(qkvh, vtAll);

    hipMemsetAsync(attn32, 0, (size_t)Tc * Dc * sizeof(float), stream);
    for (int e = 0; e < Ec; ++e) {
      const half_t* qbase = qkvh + (size_t)e * Tc * D3c;
      // ---- scores = q k^T / sqrt(HD) + causal mask   (batched over b,h)
      { GemmP p{};
        p.M = Sc; p.N = Sc; p.K = HDc;
        p.A = qbase;      p.lda = (long)Bc * D3c; p.a0 = HDc; p.a1 = D3c;
        p.B = qbase + Dc; p.ldb = (long)Bc * D3c; p.b0 = HDc; p.b1 = D3c;
        p.Z0 = Hc;
        p.alpha = 0.125f; p.causal = 1;
        p.out32 = sc32; p.ldc = Sc; p.c0 = (long)Sc * Sc; p.c1 = (long)Hc * Sc * Sc;
        launch_gemm(p, BHc, stream);
      }
      softmax_rows<<<BHc * Sc, 256, 0, stream>>>(sc32, probs, Sc);
      // ---- o = probs @ v   (vt stored [b][h][hd][s] = [N][K])
      { GemmP p{};
        p.M = Sc; p.N = HDc; p.K = Sc;
        p.A = probs; p.lda = Sc; p.a0 = (long)Sc * Sc; p.a1 = (long)Hc * Sc * Sc;
        p.B = vtAll + (size_t)e * BHc * HDc * Sc; p.ldb = Sc;
        p.b0 = (long)HDc * Sc; p.b1 = (long)Hc * HDc * Sc;
        p.Z0 = Hc;
        p.alpha = 1.f;
        p.out16 = oh; p.ldc = (long)Bc * Dc; p.c0 = HDc; p.c1 = Dc;
        launch_gemm(p, BHc, stream);
      }
      // ---- attn += g[:,e] * (o @ Wo[e]^T + bo[e])
      { GemmP p{};
        p.M = Tc; p.N = Dc; p.K = Dc;
        p.A = oh; p.lda = Dc;
        p.B = Woh + ((size_t)l * Ec + e) * Dc * Dc; p.ldb = Dc;
        p.Z0 = 1;
        p.bias = bo + ((size_t)l * Ec + e) * Dc;
        p.rowScale = g32 + e; p.rsRow = Ec;
        p.alpha = 1.f;
        p.out32 = attn32; p.accum = 1; p.ldc = Dc;
        launch_gemm(p, 1, stream);
      }
    }

    // ---- x = ln2(ln1(x + attn))
    ln_kernel<<<Tc, 256, 0, stream>>>(x32, attn32, ln1g + (size_t)l * Dc, ln1b + (size_t)l * Dc, x32, nullptr);
    ln_kernel<<<Tc, 256, 0, stream>>>(x32, nullptr, ln2g + (size_t)l * Dc, ln2b + (size_t)l * Dc, x32, xh);

    // ---- h = relu(x @ W1^T + b1)
    { GemmP p{};
      p.M = Tc; p.N = FFc; p.K = Dc;
      p.A = xh; p.lda = Dc;
      p.B = W1h + (size_t)l * FFc * Dc; p.ldb = Dc;
      p.Z0 = 1;
      p.bias = b1 + (size_t)l * FFc;
      p.alpha = 1.f; p.relu = 1;
      p.out16 = hh; p.ldc = FFc;
      launch_gemm(p, 1, stream);
    }
    // ---- ff = h @ W2^T + b2
    { GemmP p{};
      p.M = Tc; p.N = Dc; p.K = FFc;
      p.A = hh; p.lda = FFc;
      p.B = W2h + (size_t)l * Dc * FFc; p.ldb = FFc;
      p.Z0 = 1;
      p.bias = b2 + (size_t)l * Dc;
      p.alpha = 1.f;
      p.out32 = ff32; p.ldc = Dc;
      launch_gemm(p, 1, stream);
    }
    // ---- x = ln3(x + ff)
    ln_kernel<<<Tc, 256, 0, stream>>>(x32, ff32, ln3g + (size_t)l * Dc, ln3b + (size_t)l * Dc, x32, xh);
  }

  // ---- final layernorm straight into d_out
  ln_kernel<<<Tc, 256, 0, stream>>>(x32, nullptr, lnfg, lnfb, out, nullptr);
}
